// CausalSelfAttention_42082089566682
// MI455X (gfx1250) — compile-verified, matrix operand order checked by reading, not run on hardware
//
#include <hip/hip_runtime.h>
#include <hip/hip_bf16.h>

// ---------------------------------------------------------------------------
// CDNA5 (gfx1250) causal self-attention forward:
//   qkv = x @ w_attn + b_attn   (WMMA bf16 GEMM, B staged via TDM -> LDS)
//   flash-attention             (WMMA bf16, online softmax, LDS transpose)
//   out = att @ w_proj + b_proj (same GEMM)
// Weights prepacked once per launch into bf16 B-fragment order so the TDM
// can stream them into LDS as plain contiguous tiles.
// ---------------------------------------------------------------------------

typedef __attribute__((ext_vector_type(16))) __bf16 v16bf;
typedef __attribute__((ext_vector_type(8)))  __bf16 v8bf;
typedef __attribute__((ext_vector_type(8)))  float  v8f;
typedef __attribute__((ext_vector_type(4)))  unsigned int u32x4;
typedef __attribute__((ext_vector_type(8)))  int i32x8;
typedef __attribute__((ext_vector_type(4)))  int i32x4;

#define LOG2E 1.4426950408889634f

// Problem constants
#define HID   1024
#define C3    3072
#define TLEN  2048
#define NHEAD 16
#define HDIM  64
#define BATCH 2
#define MROWS 4096   // B*T
#define CHUNK 4      // kt tiles per TDM stage (4 * 1KB per N-tile row)

// ---------------------------------------------------------------------------
// Fragment loaders.
// 16-bit A/B fragment (16x32 / 32x16) per-lane element -> K mapping:
//   lane = (m_or_n & 15) + 16*hi ; element e<8 : k = 8*hi + e
//                                  element e>=8: k = 8*hi + e + 8
// Caller pre-offsets the pointer by 8*hi along K.
// ---------------------------------------------------------------------------
static __device__ __forceinline__ v16bf frag_from_f32_contig(const float* p) {
  v16bf r;
#pragma unroll
  for (int e = 0; e < 8; ++e)  r[e] = (__bf16)p[e];        // k rel = 0..7
#pragma unroll
  for (int e = 8; e < 16; ++e) r[e] = (__bf16)p[e + 8];    // k rel = 16..23
  return r;
}

static __device__ __forceinline__ v16bf frag_from_f32_strided(const float* p, int stride) {
  v16bf r;
#pragma unroll
  for (int e = 0; e < 8; ++e)  r[e] = (__bf16)p[(size_t)e * stride];
#pragma unroll
  for (int e = 8; e < 16; ++e) r[e] = (__bf16)p[(size_t)(e + 8) * stride];
  return r;
}

// Reductions within a 16-lane half-wave (matches C/D matrix row layout).
static __device__ __forceinline__ float half16_max(float v) {
  v = fmaxf(v, __shfl_xor(v, 1, 32));
  v = fmaxf(v, __shfl_xor(v, 2, 32));
  v = fmaxf(v, __shfl_xor(v, 4, 32));
  v = fmaxf(v, __shfl_xor(v, 8, 32));
  return v;
}
static __device__ __forceinline__ float half16_sum(float v) {
  v += __shfl_xor(v, 1, 32);
  v += __shfl_xor(v, 2, 32);
  v += __shfl_xor(v, 4, 32);
  v += __shfl_xor(v, 8, 32);
  return v;
}

// ---------------------------------------------------------------------------
// TDM: 2D tensor_load_to_lds. Loads tile_dim1=4 rows (N-tiles) of
// tile_dim0 = CHUNK KB contiguous bytes each, row stride = row_bytes.
// Descriptor packing per CDNA5 ISA ch.8 (D# groups 0/1; groups 2/3 unused).
// ---------------------------------------------------------------------------
static __device__ __forceinline__ void tdm_load_panel(
    unsigned long long gaddr, unsigned int lds_off,
    unsigned int row_bytes, unsigned int nrows_total) {
  const unsigned int td0 = row_bytes;          // tensor_dim0 (1B units)
  const unsigned int td1 = nrows_total;        // tensor_dim1
  const unsigned int tl0 = CHUNK * 1024u;      // tile_dim0 (bytes)
  const unsigned int tl1 = 4u;                 // tile_dim1 (rows)
  const unsigned long long s0 = row_bytes;     // tensor_dim0_stride

  u32x4 g0;
  g0[0] = 1u;                                            // count=1, user mode
  g0[1] = lds_off;                                       // lds_addr (bytes)
  g0[2] = (unsigned int)(gaddr & 0xffffffffull);         // global_addr[31:0]
  g0[3] = (unsigned int)((gaddr >> 32) & 0x01ffffffull)  // global_addr[56:32]
          | 0x80000000u;                                 // type=2 ("image")

  i32x8 g1;
  g1[0] = 0;                                             // wg_mask=0, data_size=0 (1B)
  g1[1] = (int)((td0 & 0xffffu) << 16);                  // tensor_dim0[15:0]
  g1[2] = (int)((td0 >> 16) | ((td1 & 0xffffu) << 16));  // dim0 hi | dim1 lo
  g1[3] = (int)((td1 >> 16) | (tl0 << 16));              // dim1 hi | tile_dim0
  g1[4] = (int)tl1;                                      // tile_dim1, tile_dim2=0
  g1[5] = (int)(unsigned int)(s0 & 0xffffffffull);       // dim0_stride[31:0]
  g1[6] = (int)(unsigned int)((s0 >> 32) & 0xffffu);     // dim0_stride[47:32]
  g1[7] = 0;                                             // dim1_stride (unused, 2D)

  i32x4 g2; g2[0] = 0; g2[1] = 0; g2[2] = 0; g2[3] = 0;
  i32x4 g3; g3[0] = 0; g3[1] = 0; g3[2] = 0; g3[3] = 0;

#if __clang_major__ >= 23
  i32x8 g4;
#pragma unroll
  for (int i = 0; i < 8; ++i) g4[i] = 0;
  __builtin_amdgcn_tensor_load_to_lds(g0, g1, g2, g3, g4, 0);
#else
  __builtin_amdgcn_tensor_load_to_lds(g0, g1, g2, g3, 0);
#endif
}

// ---------------------------------------------------------------------------
// Prepack: W (KxN f32 row-major) -> bf16 tiles in B-fragment order.
// Tile (kt, nt) stored at ((nt*Kt + kt)*512 + lane*16): 16 contiguous bf16
// per lane, so a B-fragment is one 32B contiguous chunk per lane.
// ---------------------------------------------------------------------------
__global__ __launch_bounds__(32) void prepack_b_kernel(
    const float* __restrict__ W, __bf16* __restrict__ out, int K, int N) {
  const int Kt   = K >> 5;
  const int tile = blockIdx.x;
  const int kt   = tile % Kt;
  const int nt   = tile / Kt;
  const int lane = threadIdx.x & 31;
  const int l15  = lane & 15;
  const int hi   = lane >> 4;
  const float* p = W + (size_t)(kt * 32 + 8 * hi) * N + nt * 16 + l15;
  v16bf v = frag_from_f32_strided(p, N);
  *(v16bf*)(out + (size_t)tile * 512 + (size_t)lane * 16) = v;
}

// ---------------------------------------------------------------------------
// GEMM: C[M,N] = A[M,K](f32 -> bf16) x Bp(prepacked bf16) + bias, f32 out.
// Block = 128 threads = 4 waves; wave w owns a 32-row strip (two A frags,
// eight 16x16 f32 accumulators over a 64-wide N panel). B panel staged into
// LDS by the Tensor Data Mover (double buffered, wave 0 issues + waits).
// ---------------------------------------------------------------------------
__global__ __launch_bounds__(128) void gemm_bias_wmma_kernel(
    const float* __restrict__ A, const __bf16* __restrict__ Bp,
    const float* __restrict__ bias, float* __restrict__ C,
    int M, int N, int K) {
  const int lane = threadIdx.x & 31;
  const int wave = threadIdx.x >> 5;
  const int l15  = lane & 15;
  const int hi   = lane >> 4;
  const int Mb   = blockIdx.y * 128 + wave * 32;
  const int Nb   = blockIdx.x * 64;
  const int Kt   = K >> 5;
  const int NC   = Kt / CHUNK;

  // Double-buffered B panel: [buf][ (t*CHUNK + kc)*512 + lane*16 ]
  __shared__ __align__(16) __bf16 bpanel[2][4 * CHUNK * 512];   // 2 x 16 KB

  v8f acc[2][4];
#pragma unroll
  for (int u = 0; u < 2; ++u)
#pragma unroll
    for (int t = 0; t < 4; ++t)
#pragma unroll
      for (int i = 0; i < 8; ++i) acc[u][t][i] = 0.0f;

  const float* arow0 = A + (size_t)(Mb + l15) * K + 8 * hi;
  const float* arow1 = arow0 + (size_t)16 * K;

  const unsigned long long bglobal =
      (unsigned long long)(uintptr_t)Bp +
      (unsigned long long)((size_t)(Nb >> 4) * Kt) * 1024ull;
  const unsigned int row_bytes = (unsigned int)Kt * 1024u;   // one N-tile row
  const unsigned int ntiles_n  = (unsigned int)(N >> 4);

  if (wave == 0)
    tdm_load_panel(bglobal, (unsigned int)(uintptr_t)&bpanel[0][0],
                   row_bytes, ntiles_n);

  int buf = 0;
  for (int c = 0; c < NC; ++c) {
    if (wave == 0) {
      if (c + 1 < NC) {
        tdm_load_panel(bglobal + (unsigned long long)(c + 1) * (CHUNK * 1024ull),
                       (unsigned int)(uintptr_t)&bpanel[buf ^ 1][0],
                       row_bytes, ntiles_n);
        __builtin_amdgcn_s_wait_tensorcnt((short)1);   // chunk c landed
      } else {
        __builtin_amdgcn_s_wait_tensorcnt((short)0);   // last chunk landed
      }
    }
    __syncthreads();   // B chunk visible to all waves

#pragma unroll
    for (int kc = 0; kc < CHUNK; ++kc) {
      const int kt = c * CHUNK + kc;
      v16bf af0 = frag_from_f32_contig(arow0 + kt * 32);
      v16bf af1 = frag_from_f32_contig(arow1 + kt * 32);
#pragma unroll
      for (int t = 0; t < 4; ++t) {
        v16bf bf_ = *(const v16bf*)(&bpanel[buf][(size_t)(t * CHUNK + kc) * 512] +
                                    (size_t)lane * 16);
        acc[0][t] = __builtin_amdgcn_wmma_f32_16x16x32_bf16(
            false, af0, false, bf_, (short)0, acc[0][t], false, false);
        acc[1][t] = __builtin_amdgcn_wmma_f32_16x16x32_bf16(
            false, af1, false, bf_, (short)0, acc[1][t], false, false);
      }
    }
    __syncthreads();   // done reading buf before TDM overwrites it
    buf ^= 1;
  }

#pragma unroll
  for (int t = 0; t < 4; ++t) {
    const int n  = Nb + t * 16 + l15;
    const float bv = bias[n];
#pragma unroll
    for (int u = 0; u < 2; ++u)
#pragma unroll
      for (int r = 0; r < 8; ++r) {
        const int m = Mb + u * 16 + r + 8 * hi;
        C[(size_t)m * N + n] = acc[u][t][r] + bv;
      }
  }
}

// ---------------------------------------------------------------------------
// Flash attention: one wave per (b, head, 16-query tile). Streams 32-key
// tiles with online softmax; P is transposed C-layout -> A-fragment via LDS.
// ---------------------------------------------------------------------------
__global__ __launch_bounds__(32) void attn_wmma_kernel(
    const float* __restrict__ qkv, float* __restrict__ attout) {
  const int lane = threadIdx.x & 31;
  const int l15  = lane & 15;
  const int hi   = lane >> 4;

  int idx = blockIdx.x;
  const int qt = idx & 127;  idx >>= 7;   // T/16 = 128 query tiles
  const int h  = idx & 15;   idx >>= 4;
  const int b  = idx;                      // 0..1
  const int qbase = qt * 16;
  const int Hoff  = h * HDIM;

  // Q fragments for d=0..31 and d=32..63 (loaded once).
  const size_t rowq = (size_t)(b * TLEN + qbase + l15) * C3;
  const v16bf q0 = frag_from_f32_contig(qkv + rowq + Hoff + 0  + 8 * hi);
  const v16bf q1 = frag_from_f32_contig(qkv + rowq + Hoff + 32 + 8 * hi);

  v8f o[4];
  float m_run[8], l_run[8];
#pragma unroll
  for (int t = 0; t < 4; ++t)
#pragma unroll
    for (int i = 0; i < 8; ++i) o[t][i] = 0.0f;
#pragma unroll
  for (int r = 0; r < 8; ++r) { m_run[r] = -3.0e38f; l_run[r] = 0.0f; }

  __shared__ __align__(16) __bf16 plds[16 * 32];   // 1 KB P tile (bf16)

  const int njt = (qbase + 47) >> 5;   // #32-key tiles covering keys <= qbase+15
  for (int jt = 0; jt < njt; ++jt) {
    const int jbase = jt * 32;

    // --- S = Q @ K^T over d=64 (two K-steps), two 16-key accumulators ---
    const float* kp = qkv + (size_t)(b * TLEN + jbase) * C3 + HID + Hoff + 8 * hi;
    const v16bf k00 = frag_from_f32_contig(kp + (size_t)l15 * C3);
    const v16bf k01 = frag_from_f32_contig(kp + (size_t)l15 * C3 + 32);
    const v16bf k10 = frag_from_f32_contig(kp + (size_t)(16 + l15) * C3);
    const v16bf k11 = frag_from_f32_contig(kp + (size_t)(16 + l15) * C3 + 32);

    v8f s0, s1;
#pragma unroll
    for (int i = 0; i < 8; ++i) { s0[i] = 0.0f; s1[i] = 0.0f; }
    s0 = __builtin_amdgcn_wmma_f32_16x16x32_bf16(false, q0, false, k00, (short)0, s0, false, false);
    s0 = __builtin_amdgcn_wmma_f32_16x16x32_bf16(false, q1, false, k01, (short)0, s0, false, false);
    s1 = __builtin_amdgcn_wmma_f32_16x16x32_bf16(false, q0, false, k10, (short)0, s1, false, false);
    s1 = __builtin_amdgcn_wmma_f32_16x16x32_bf16(false, q1, false, k11, (short)0, s1, false, false);

    // --- scale, causal mask, online softmax (per row r+8*hi) ---
    float p0[8], p1[8];
#pragma unroll
    for (int r = 0; r < 8; ++r) {
      const int qrow = qbase + r + 8 * hi;
      float v0 = s0[r] * 0.125f;                 // 1/sqrt(64)
      float v1 = s1[r] * 0.125f;
      if (jbase + l15 > qrow)      v0 = -3.0e38f;
      if (jbase + 16 + l15 > qrow) v1 = -3.0e38f;

      float mt = half16_max(fmaxf(v0, v1));
      const float mn = fmaxf(m_run[r], mt);
      const float al = exp2f((m_run[r] - mn) * LOG2E);
      p0[r] = exp2f((v0 - mn) * LOG2E);
      p1[r] = exp2f((v1 - mn) * LOG2E);
      const float rs = half16_sum(p0[r] + p1[r]);
      l_run[r] = l_run[r] * al + rs;
      m_run[r] = mn;
#pragma unroll
      for (int t = 0; t < 4; ++t) o[t][r] *= al;
    }

    // --- transpose P (C-layout) -> A-fragment via LDS (row-major 16x32 bf16)
#pragma unroll
    for (int r = 0; r < 8; ++r) {
      const int m = r + 8 * hi;
      plds[m * 32 + l15]      = (__bf16)p0[r];
      plds[m * 32 + 16 + l15] = (__bf16)p1[r];
    }
    __syncthreads();
    const v8bf* lp = (const v8bf*)(plds + l15 * 32 + 8 * hi);
    const v8bf plo = lp[0];          // keys 8*hi + 0..7
    const v8bf phi = lp[2];          // keys 8*hi + 16..23
    const v16bf pa = __builtin_shufflevector(plo, phi,
        0, 1, 2, 3, 4, 5, 6, 7, 8, 9, 10, 11, 12, 13, 14, 15);
    __syncthreads();

    // --- O += P @ V  (K = 32 keys, four 16-wide d tiles) ---
    const float* vp = qkv + (size_t)(b * TLEN + jbase + 8 * hi) * C3
                      + 2 * HID + Hoff + l15;
#pragma unroll
    for (int t = 0; t < 4; ++t) {
      v16bf vf = frag_from_f32_strided(vp + t * 16, C3);
      o[t] = __builtin_amdgcn_wmma_f32_16x16x32_bf16(
          false, pa, false, vf, (short)0, o[t], false, false);
    }
  }

  // --- normalize and write out (B,T,C layout) ---
#pragma unroll
  for (int r = 0; r < 8; ++r) {
    const float inv = 1.0f / l_run[r];
    const int   m   = qbase + r + 8 * hi;
    const size_t rowo = (size_t)(b * TLEN + m) * HID + Hoff;
#pragma unroll
    for (int t = 0; t < 4; ++t)
      attout[rowo + t * 16 + l15] = o[t][r] * inv;
  }
}

// ---------------------------------------------------------------------------
// Host-side launch
// ---------------------------------------------------------------------------
extern "C" void kernel_launch(void* const* d_in, const int* in_sizes, int n_in,
                              void* d_out, int out_size, void* d_ws, size_t ws_size,
                              hipStream_t stream) {
  const float* x      = (const float*)d_in[0];
  const float* w_attn = (const float*)d_in[1];
  const float* b_attn = (const float*)d_in[2];
  const float* w_proj = (const float*)d_in[3];
  const float* b_proj = (const float*)d_in[4];
  float* out = (float*)d_out;

  char* ws = (char*)d_ws;
  const size_t QKV_BYTES = (size_t)MROWS * C3  * sizeof(float);  // 48 MB
  const size_t ATT_BYTES = (size_t)MROWS * HID * sizeof(float);  // 16 MB
  const size_t WBA_BYTES = (size_t)HID * C3 * sizeof(__bf16);    //  6 MB

  float*  qkv    = (float*)(ws);
  float*  attout = (float*)(ws + QKV_BYTES);
  __bf16* wbA    = (__bf16*)(ws + QKV_BYTES + ATT_BYTES);
  __bf16* wbP    = (__bf16*)(ws + QKV_BYTES + ATT_BYTES + WBA_BYTES);

  // Prepack weights into bf16 B-fragment tiles.
  prepack_b_kernel<<<(HID / 32) * (C3 / 16),  32, 0, stream>>>(w_attn, wbA, HID, C3);
  prepack_b_kernel<<<(HID / 32) * (HID / 16), 32, 0, stream>>>(w_proj, wbP, HID, HID);

  // qkv = x @ w_attn + b_attn
  gemm_bias_wmma_kernel<<<dim3(C3 / 64, MROWS / 128), 128, 0, stream>>>(
      x, wbA, b_attn, qkv, MROWS, C3, HID);

  // causal flash attention: B * NHEAD * (T/16) waves
  attn_wmma_kernel<<<BATCH * NHEAD * (TLEN / 16), 32, 0, stream>>>(qkv, attout);

  // out = attout @ w_proj + b_proj
  gemm_bias_wmma_kernel<<<dim3(HID / 64, MROWS / 128), 128, 0, stream>>>(
      attout, wbP, b_proj, out, MROWS, HID, HID);
}
